// DeformConv_15796889715251
// MI455X (gfx1250) — compile-verified, matrix operand order checked by reading, not run on hardware
//
#include <hip/hip_runtime.h>
#include <hip/hip_bf16.h>

typedef unsigned int  uint;
typedef unsigned short ushort;
typedef __attribute__((ext_vector_type(16))) __bf16 v16bf;
typedef __attribute__((ext_vector_type(8)))  float  v8f;
typedef __attribute__((ext_vector_type(4)))  int    i4v;

#define Hn   128
#define Wn   128
#define HW   16384      // Hn*Wn
#define Cin  256
#define CoutN 256
#define Kpts 9
#define KD   2304       // Cin*Kpts
#define Bn   2
#define NPIX 32768      // Bn*HW
#define TILE_M 64
#define TILE_N 256      // full Cout in one tile: halves total gather work
#define TILE_K 32

#if defined(__has_builtin)
#if __has_builtin(__builtin_amdgcn_global_load_async_to_lds_b128) && \
    __has_builtin(__builtin_amdgcn_s_wait_asynccnt)
#define USE_ASYNC_LDS 1
#endif
#endif

typedef __attribute__((address_space(1))) i4v* gptr_t;   // global int4*
typedef __attribute__((address_space(3))) i4v* lptr_t;   // LDS int4*

__device__ __forceinline__ ushort f2bf(float f) {
    uint u = __float_as_uint(f);
    uint r = u + 0x7FFFu + ((u >> 16) & 1u);   // round-to-nearest-even
    return (ushort)(r >> 16);
}

// ---------------------------------------------------------------------------
// Kernel 1: 27-channel 3x3 offset conv + bilinear coefficient table.
// One thread per pixel. coefW = 4 corner weights (mask & validity folded in),
// coefI = 4 clamped linear offsets within one (b,c) plane.
// ---------------------------------------------------------------------------
__global__ __launch_bounds__(256) void offset_coef_kernel(
    const float* __restrict__ x, const float* __restrict__ w_off,
    const float* __restrict__ b_off,
    float4* __restrict__ coefW, int4* __restrict__ coefI)
{
    int p = blockIdx.x * 256 + threadIdx.x;          // 0..NPIX-1
    int b = p >> 14, hw = p & (HW - 1);
    int h = hw >> 7, w = hw & (Wn - 1);

    float acc[27];
    #pragma unroll
    for (int j = 0; j < 27; ++j) acc[j] = b_off[j];

    const float* xb = x + (size_t)b * Cin * HW;
    for (int c = 0; c < Cin; ++c) {
        float v[9];
        #pragma unroll
        for (int dy = 0; dy < 3; ++dy)
            #pragma unroll
            for (int dx = 0; dx < 3; ++dx) {
                int yy = h + dy - 1, xx = w + dx - 1;
                bool ok = (yy >= 0) & (yy < Hn) & (xx >= 0) & (xx < Wn);
                v[dy * 3 + dx] = ok ? xb[c * HW + yy * Wn + xx] : 0.f;
            }
        const float* wc = w_off + c * 9;             // w_off[j][c][i]
        #pragma unroll
        for (int j = 0; j < 27; ++j) {
            const float* wj = wc + j * (Cin * 9);
            float a = acc[j];
            #pragma unroll
            for (int i = 0; i < 9; ++i) a = fmaf(v[i], wj[i], a);
            acc[j] = a;
        }
    }

    #pragma unroll
    for (int k = 0; k < Kpts; ++k) {
        float oy = acc[k], ox = acc[9 + k];
        float mk = 1.f / (1.f + __expf(-acc[18 + k]));
        float py = (float)h - 1.f + (float)(k / 3) + oy;
        float px = (float)w - 1.f + (float)(k % 3) + ox;
        float y0f = floorf(py), x0f = floorf(px);
        float wy1 = py - y0f, wx1 = px - x0f;
        int y0 = (int)y0f, x0 = (int)x0f;
        bool vy0 = (y0 >= 0) & (y0 < Hn), vy1 = (y0 + 1 >= 0) & (y0 + 1 < Hn);
        bool vx0 = (x0 >= 0) & (x0 < Wn), vx1 = (x0 + 1 >= 0) & (x0 + 1 < Wn);
        float4 cw;
        cw.x = (vy0 && vx0) ? (1.f - wy1) * (1.f - wx1) * mk : 0.f;
        cw.y = (vy0 && vx1) ? (1.f - wy1) * wx1 * mk : 0.f;
        cw.z = (vy1 && vx0) ? wy1 * (1.f - wx1) * mk : 0.f;
        cw.w = (vy1 && vx1) ? wy1 * wx1 * mk : 0.f;
        int y0c = min(max(y0, 0), Hn - 1), y1c = min(max(y0 + 1, 0), Hn - 1);
        int x0c = min(max(x0, 0), Wn - 1), x1c = min(max(x0 + 1, 0), Wn - 1);
        int4 ci = { y0c * Wn + x0c, y0c * Wn + x1c, y1c * Wn + x0c, y1c * Wn + x1c };
        coefW[p * Kpts + k] = cw;
        coefI[p * Kpts + k] = ci;
    }
}

// ---------------------------------------------------------------------------
// Kernel 2: convert weights fp32 -> bf16.  weight (Cout,C,3,3) flat is already
// [o][c*9+k] = [o][kd] order -> B operand rows.
// ---------------------------------------------------------------------------
__global__ __launch_bounds__(256) void wconvert_kernel(
    const float* __restrict__ w, ushort* __restrict__ wb, int n)
{
    int i = blockIdx.x * 256 + threadIdx.x;
    if (i < n) wb[i] = f2bf(w[i]);
}

// ---------------------------------------------------------------------------
// Kernel 3: fused deformable-gather + bf16 WMMA GEMM + bias + stats partials.
// Block = 256 threads = 8 waves, tile 64(M=pixels) x 256(N = all out chans).
// Wave (wm,wn): wm in {0,1} covers 32 rows (2 A-frags), wn in {0..3} covers
// 64 cols (4 B-frags) -> 2x4 v_wmma_f32_16x16x32_bf16 per K-step per wave.
// B tile staged with GLOBAL_LOAD_ASYNC_TO_LDS (ASYNCcnt) when available.
// ---------------------------------------------------------------------------
__global__ __launch_bounds__(256) void deform_gemm_kernel(
    const float* __restrict__ x,
    const float4* __restrict__ coefW,
    const int4*  __restrict__ coefI,
    const ushort* __restrict__ Wb,     // [Cout][KD] bf16
    const float* __restrict__ bias,
    float* __restrict__ y,             // [B][Cout][H][W]
    float* __restrict__ gsum, float* __restrict__ gsum2)
{
    __shared__ float4 sCW[TILE_M * Kpts];        //  9216 B
    __shared__ int4   sCI[TILE_M * Kpts];        //  9216 B
    __shared__ uint   sA[TILE_M][TILE_K / 2];    //  4096 B (bf16 pairs)
    __shared__ uint   sB[TILE_N][TILE_K / 2];    // 16384 B
    __shared__ float  sY[TILE_M * 65];           // 16640 B (64x64 chunk, padded)
    __shared__ float  sSum[64], sSum2[64];       //   512 B
                                                 // total: 56064 B

    const int tid  = threadIdx.x;
    const int lane = tid & 31;
    const int wave = tid >> 5;
    const int wm = wave & 1;          // M group (2 x 32 rows)
    const int wn = wave >> 1;         // N group (4 x 64 cols)
    const int l16   = lane & 15;
    const int khalf = lane >> 4;      // K-half select per ISA A/B/C layouts

    const int p0 = blockIdx.x * TILE_M;           // pixel tile base
    const int bIdx = p0 >> 14;                    // whole tile in one batch
    const int hw0  = p0 & (HW - 1);
    const float* xb = x + (size_t)bIdx * Cin * HW;

    // coefficient tile (contiguous in global): reused across all 72 K-steps
    for (int i = tid; i < TILE_M * Kpts; i += 256) {
        sCW[i] = coefW[p0 * Kpts + i];
        sCI[i] = coefI[p0 * Kpts + i];
    }
    __syncthreads();

    v8f acc[2][4];
    #pragma unroll
    for (int i = 0; i < 2; ++i)
        #pragma unroll
        for (int j = 0; j < 4; ++j) acc[i][j] = (v8f){};

    const int mLoc   = tid & 63;       // A-build: pixel row
    const int kgroup = tid >> 6;       // A-build: 8 consecutive kk

    for (int kd0 = 0; kd0 < KD; kd0 += TILE_K) {
        // ---- stage B tile: thread t copies row t (32 bf16 = 64 B) ----
#ifdef USE_ASYNC_LDS
        {
            gptr_t gp = (gptr_t)(Wb + (size_t)tid * KD + kd0);   // 16B-aligned
            lptr_t lp = (lptr_t)&sB[tid][0];                     // 64B rows
            __builtin_amdgcn_global_load_async_to_lds_b128(gp,     lp,     0, 0);
            __builtin_amdgcn_global_load_async_to_lds_b128(gp + 1, lp + 1, 0, 0);
            __builtin_amdgcn_global_load_async_to_lds_b128(gp + 2, lp + 2, 0, 0);
            __builtin_amdgcn_global_load_async_to_lds_b128(gp + 3, lp + 3, 0, 0);
        }
#else
        {
            const uint4* wp = (const uint4*)(Wb + (size_t)tid * KD + kd0);
            uint4 b0 = wp[0], b1 = wp[1], b2 = wp[2], b3 = wp[3];
            uint4* dst = (uint4*)&sB[tid][0];
            dst[0] = b0; dst[1] = b1; dst[2] = b2; dst[3] = b3;
        }
#endif
        // ---- build A tile (gather + bilinear + bf16 pack) ----
        #pragma unroll
        for (int i = 0; i < 4; ++i) {
            uint pk = 0;
            #pragma unroll
            for (int h2 = 0; h2 < 2; ++h2) {
                int kk = (kgroup << 3) + (i << 1) + h2;
                int kd = kd0 + kk;
                int c  = kd / 9;
                int k  = kd - c * 9;
                float4 wg = sCW[mLoc * Kpts + k];
                int4   ci = sCI[mLoc * Kpts + k];
                const float* xp = xb + c * HW;       // L2-resident gathers
                float v = wg.x * xp[ci.x] + wg.y * xp[ci.y]
                        + wg.z * xp[ci.z] + wg.w * xp[ci.w];
                pk |= (uint)f2bf(v) << (h2 * 16);
            }
            sA[mLoc][(kgroup << 2) + i] = pk;
        }
#ifdef USE_ASYNC_LDS
        __builtin_amdgcn_s_wait_asynccnt(0);
#endif
        __syncthreads();

        // ---- fragments + WMMA ----
        union FragU { v16bf v; uint4 q[2]; };
        FragU afr[2], bfr[4];
        #pragma unroll
        for (int s = 0; s < 2; ++s) {
            int m = wm * 32 + s * 16 + l16;
            afr[s].q[0] = *(const uint4*)&sA[m][khalf * 4];       // K 0..7 / 8..15
            afr[s].q[1] = *(const uint4*)&sA[m][8 + khalf * 4];   // K 16..23 / 24..31
        }
        #pragma unroll
        for (int s = 0; s < 4; ++s) {
            int n = wn * 64 + s * 16 + l16;
            bfr[s].q[0] = *(const uint4*)&sB[n][khalf * 4];
            bfr[s].q[1] = *(const uint4*)&sB[n][8 + khalf * 4];
        }
        #pragma unroll
        for (int i = 0; i < 2; ++i)
            #pragma unroll
            for (int j = 0; j < 4; ++j)
                acc[i][j] = __builtin_amdgcn_wmma_f32_16x16x32_bf16(
                    false, afr[i].v, false, bfr[j].v,
                    (short)0, acc[i][j], false, false);
        __syncthreads();
    }

    // ---- epilogue: 4 passes over 64-channel chunks through one sY buffer ----
    for (int ch = 0; ch < 4; ++ch) {
        if (tid < 64) { sSum[tid] = 0.f; sSum2[tid] = 0.f; }
        if (wn == ch) {
            #pragma unroll
            for (int j = 0; j < 4; ++j) {
                int nl = j * 16 + l16;            // col within chunk
                float bo = bias[ch * 64 + nl];
                #pragma unroll
                for (int i = 0; i < 2; ++i)
                    #pragma unroll
                    for (int r = 0; r < 8; ++r) { // C/D layout: M = r + 8*khalf
                        int m = wm * 32 + i * 16 + khalf * 8 + r;
                        sY[m * 65 + nl] = acc[i][j][r] + bo;
                    }
            }
        }
        __syncthreads();
        // per-channel partial sums for batchnorm
        {
            int col = tid & 63;
            int rbase = (tid >> 6) * 16;
            float s = 0.f, s2 = 0.f;
            #pragma unroll 4
            for (int r = 0; r < 16; ++r) {
                float v = sY[(rbase + r) * 65 + col];
                s += v; s2 += v * v;
            }
            atomicAdd(&sSum[col], s);
            atomicAdd(&sSum2[col], s2);
        }
        __syncthreads();
        if (tid < 64) {
            atomicAdd(&gsum[ch * 64 + tid],  sSum[tid]);
            atomicAdd(&gsum2[ch * 64 + tid], sSum2[tid]);
        }
        // coalesced y store: consecutive lanes -> consecutive hw
        for (int e = tid; e < 64 * 64; e += 256) {
            int n = e >> 6, m = e & 63;
            y[(size_t)(bIdx * CoutN + ch * 64 + n) * HW + hw0 + m] = sY[m * 65 + n];
        }
        __syncthreads();
    }
}

// ---------------------------------------------------------------------------
// Kernel 4: fold sums into scale/shift.
// ---------------------------------------------------------------------------
__global__ void stats_kernel(const float* __restrict__ gsum, const float* __restrict__ gsum2,
                             const float* __restrict__ gamma, const float* __restrict__ beta,
                             float* __restrict__ scale, float* __restrict__ shift)
{
    int o = threadIdx.x;
    if (o < CoutN) {
        const float invN = 1.f / (float)NPIX;
        float mu  = gsum[o] * invN;
        float var = gsum2[o] * invN - mu * mu;
        float rs  = rsqrtf(var + 1e-5f);
        float sc  = gamma[o] * rs;
        scale[o] = sc;
        shift[o] = beta[o] - mu * sc;
    }
}

// ---------------------------------------------------------------------------
// Kernel 5: normalize + affine + ReLU.
// ---------------------------------------------------------------------------
__global__ __launch_bounds__(256) void finalize_kernel(
    const float* __restrict__ y, const float* __restrict__ scale,
    const float* __restrict__ shift, float* __restrict__ out)
{
    int i = blockIdx.x * 256 + threadIdx.x;
    int o = (i >> 14) & (CoutN - 1);
    float v = fmaf(y[i], scale[o], shift[o]);
    out[i] = v > 0.f ? v : 0.f;
}

// ---------------------------------------------------------------------------
extern "C" void kernel_launch(void* const* d_in, const int* in_sizes, int n_in,
                              void* d_out, int out_size, void* d_ws, size_t ws_size,
                              hipStream_t stream) {
    const float* x      = (const float*)d_in[0];
    const float* w_off  = (const float*)d_in[1];
    const float* b_off  = (const float*)d_in[2];
    const float* weight = (const float*)d_in[3];
    const float* bias   = (const float*)d_in[4];
    const float* gamma  = (const float*)d_in[5];
    const float* beta   = (const float*)d_in[6];
    float* out = (float*)d_out;

    char* ws = (char*)d_ws;
    // workspace layout (~44.2 MB)
    size_t o_y   = 0;                              // 33,554,432 B  y (B,Cout,H,W)
    size_t o_cw  = o_y  + (size_t)Bn * CoutN * HW * 4;
    size_t o_ci  = o_cw + (size_t)NPIX * Kpts * 16;
    size_t o_wb  = o_ci + (size_t)NPIX * Kpts * 16;
    size_t o_gs  = o_wb + (size_t)CoutN * KD * 2;
    size_t o_gs2 = o_gs  + 1024;
    size_t o_sc  = o_gs2 + 1024;
    size_t o_sh  = o_sc  + 1024;
    (void)o_sh; (void)ws_size; (void)in_sizes; (void)n_in; (void)out_size;

    float*  yb    = (float*)(ws + o_y);
    float4* coefW = (float4*)(ws + o_cw);
    int4*   coefI = (int4*)(ws + o_ci);
    ushort* Wb    = (ushort*)(ws + o_wb);
    float*  gsum  = (float*)(ws + o_gs);
    float*  gsum2 = (float*)(ws + o_gs2);
    float*  scale = (float*)(ws + o_sc);
    float*  shift = (float*)(ws + o_sh);

    // offsets + bilinear coefficients
    offset_coef_kernel<<<NPIX / 256, 256, 0, stream>>>(x, w_off, b_off, coefW, coefI);
    // bf16 weights
    wconvert_kernel<<<(CoutN * KD + 255) / 256, 256, 0, stream>>>(weight, Wb, CoutN * KD);
    // zero stats accumulators (graph-capture safe)
    (void)hipMemsetAsync(ws + o_gs, 0, 2048, stream);
    // fused gather + WMMA GEMM (one N tile: each A element gathered once)
    deform_gemm_kernel<<<dim3(NPIX / TILE_M), 256, 0, stream>>>(
        x, coefW, coefI, Wb, bias, yb, gsum, gsum2);
    // batchnorm fold
    stats_kernel<<<1, 256, 0, stream>>>(gsum, gsum2, gamma, beta, scale, shift);
    // normalize + relu
    finalize_kernel<<<(Bn * CoutN * HW) / 256, 256, 0, stream>>>(yb, scale, shift, out);
}
